// GCNConv_19645180412416
// MI455X (gfx1250) — compile-verified, matrix operand order checked by reading, not run on hardware
//
#include <hip/hip_runtime.h>

// ---------------- problem constants (match reference) ----------------
constexpr int N_NODES = 8192;
constexpr int N_EDGES = 262144;
constexpr int IN_C    = 512;
constexpr int OUT_C   = 512;
constexpr int WORDS_PER_ROW = N_NODES / 32;   // 256 u32 words of bitmask per row

// ---------------- WMMA types ----------------
typedef __attribute__((ext_vector_type(16))) __bf16 v16bf;
typedef __attribute__((ext_vector_type(8)))  float  v8f;

union Frag {
    v16bf v;
    unsigned int u[8];
};

// ---------------- helpers ----------------
__device__ __forceinline__ unsigned short f32_to_bf16_rne(float f) {
    unsigned int u = __float_as_uint(f);
    u += 0x7FFFu + ((u >> 16) & 1u);   // round-to-nearest-even
    return (unsigned short)(u >> 16);
}
__device__ __forceinline__ float bf16_to_f32(unsigned short h) {
    return __uint_as_float(((unsigned int)h) << 16);
}

// ---------------- kernel 1: zero the adjacency bitmask ----------------
__global__ __launch_bounds__(256) void zero_bitmask_kernel(uint4* bm4) {
    // 8192*256 u32 = 524288 uint4 -> exactly 2048 blocks * 256 threads
    int idx = blockIdx.x * 256 + threadIdx.x;
    bm4[idx] = uint4{0u, 0u, 0u, 0u};
}

// ---------------- kernel 2: scatter edges (+ self loops) into bitmask ----------------
__global__ __launch_bounds__(256) void set_edges_kernel(const long long* __restrict__ ei,
                                                        unsigned int* __restrict__ bm) {
    int idx = blockIdx.x * 256 + threadIdx.x;
    int total = N_EDGES + N_NODES;
    if (idx >= total) return;
    int r, c;
    if (idx < N_EDGES) {
        r = (int)ei[idx];            // edge_index[0][idx]
        c = (int)ei[N_EDGES + idx];  // edge_index[1][idx]
    } else {
        r = idx - N_EDGES;           // self loop
        c = r;
    }
    atomicOr(&bm[r * WORDS_PER_ROW + (c >> 5)], 1u << (c & 31));
}

// ---------------- kernel 3: degree popcount -> deg^{-1/2} ----------------
__global__ __launch_bounds__(256) void degree_kernel(const unsigned int* __restrict__ bm,
                                                     float* __restrict__ dinv) {
    int i = blockIdx.x * 256 + threadIdx.x;
    if (i >= N_NODES) return;
    int cnt = 0;
    const unsigned int* row = bm + i * WORDS_PER_ROW;
#pragma unroll 8
    for (int w = 0; w < WORDS_PER_ROW; ++w) cnt += __popc(row[w]);
    dinv[i] = (cnt > 0) ? rsqrtf((float)cnt) : 0.0f;
}

// ---------------- kernel 4: split f32 -> bf16 hi + bf16 lo ----------------
__global__ __launch_bounds__(256) void split_bf16_kernel(const float* __restrict__ src,
                                                         unsigned short* __restrict__ hi,
                                                         unsigned short* __restrict__ lo,
                                                         int n) {
    int idx = blockIdx.x * 256 + threadIdx.x;
    if (idx >= n) return;
    float v = src[idx];
    unsigned short h = f32_to_bf16_rne(v);
    hi[idx] = h;
    lo[idx] = f32_to_bf16_rne(v - bf16_to_f32(h));
}

// ---------------- kernel 5: WMMA GEMM  hs = dinv * (x @ W^T + b) ----------------
// Each wave computes a 16(M) x 64(N) tile. 8 waves per 256-thread block.
// A-frag (16x32 bf16) and B-frag (32x16 bf16) per-lane layouts are transposes of
// each other, so both load packed K-pairs straight from row-major hi/lo arrays.
__global__ __launch_bounds__(256) void gemm_wmma_kernel(const unsigned short* __restrict__ xhi,
                                                        const unsigned short* __restrict__ xlo,
                                                        const unsigned short* __restrict__ whi,
                                                        const unsigned short* __restrict__ wlo,
                                                        const float* __restrict__ bias,
                                                        const float* __restrict__ dinv,
                                                        float* __restrict__ hs) {
    const int lane  = threadIdx.x & 31;
    const int wave  = threadIdx.x >> 5;
    const int t     = blockIdx.x * 8 + wave;   // 512 blocks * 8 waves = 4096 wave-tiles
    const int tileM = t >> 3;                  // 0..511  (M tile of 16 rows)
    const int nBase = (t & 7) * 64;            // 0..448  (N group of 64 cols)
    const int half  = lane >> 4;               // K half selector
    const int id16  = lane & 15;               // M (for A) / N (for B) within tile

    const int rowA = tileM * 16 + id16;

    v8f acc[4] = {};

    const unsigned short* pxh = xhi + rowA * IN_C + half * 8;
    const unsigned short* pxl = xlo + rowA * IN_C + half * 8;

#pragma unroll 4
    for (int kk = 0; kk < IN_C; kk += 32) {
        Frag ah, al;
#pragma unroll
        for (int r = 0; r < 8; ++r) {
            const int off = kk + (r >> 2) * 16 + (r & 3) * 2;  // even -> dword aligned
            ah.u[r] = *(const unsigned int*)(pxh + off);
            al.u[r] = *(const unsigned int*)(pxl + off);
        }
#pragma unroll
        for (int s = 0; s < 4; ++s) {
            const int colB = nBase + s * 16 + id16;            // W row = output channel
            const unsigned short* pwh = whi + colB * IN_C + half * 8;
            const unsigned short* pwl = wlo + colB * IN_C + half * 8;
            Frag bh, bl;
#pragma unroll
            for (int r = 0; r < 8; ++r) {
                const int off = kk + (r >> 2) * 16 + (r & 3) * 2;
                bh.u[r] = *(const unsigned int*)(pwh + off);
                bl.u[r] = *(const unsigned int*)(pwl + off);
            }
            // split-f32 product: hi*hi + hi*lo + lo*hi  (lo*lo dropped)
            acc[s] = __builtin_amdgcn_wmma_f32_16x16x32_bf16(false, ah.v, false, bh.v,
                                                             (short)0, acc[s], false, false);
            acc[s] = __builtin_amdgcn_wmma_f32_16x16x32_bf16(false, ah.v, false, bl.v,
                                                             (short)0, acc[s], false, false);
            acc[s] = __builtin_amdgcn_wmma_f32_16x16x32_bf16(false, al.v, false, bh.v,
                                                             (short)0, acc[s], false, false);
        }
    }

    // Epilogue: D layout lane -> (n = id16, m = r + 8*half). Fuse bias + dinv scale.
#pragma unroll
    for (int s = 0; s < 4; ++s) {
        const int n  = nBase + s * 16 + id16;
        const float bn = bias[n];
#pragma unroll
        for (int r = 0; r < 8; ++r) {
            const int m = tileM * 16 + r + half * 8;
            hs[m * OUT_C + n] = dinv[m] * (acc[s][r] + bn);
        }
    }
}

// ---------------- kernel 6: bitmask-driven aggregation ----------------
// out[i] = dinv[i] * sum_{j : adj[i][j]} hs[j].  One block per row; hs is L2-resident.
__global__ __launch_bounds__(256) void aggregate_kernel(const unsigned int* __restrict__ bm,
                                                        const float* __restrict__ hs,
                                                        const float* __restrict__ dinv,
                                                        float* __restrict__ out) {
    __shared__ unsigned int wbuf[WORDS_PER_ROW];
    const int row = blockIdx.x;
    const int tid = threadIdx.x;

    wbuf[tid] = bm[row * WORDS_PER_ROW + tid];   // 256 threads load 256 words, coalesced
    __syncthreads();

    const int c0 = tid * 2;                      // 2 channels per thread
    float ax = 0.0f, ay = 0.0f;

    for (int w = 0; w < WORDS_PER_ROW; ++w) {
        unsigned int word = wbuf[w];
        while (word) {
            const int j = w * 32 + __builtin_ctz(word);
            word &= word - 1;
            const float2 hv = *(const float2*)(hs + j * OUT_C + c0);  // coalesced across block
            ax += hv.x;
            ay += hv.y;
        }
    }
    const float s = dinv[row];
    float2 res;
    res.x = s * ax;
    res.y = s * ay;
    *(float2*)(out + row * OUT_C + c0) = res;
}

// ---------------- host launch ----------------
extern "C" void kernel_launch(void* const* d_in, const int* in_sizes, int n_in,
                              void* d_out, int out_size, void* d_ws, size_t ws_size,
                              hipStream_t stream) {
    const float*     x  = (const float*)d_in[0];
    const long long* ei = (const long long*)d_in[1];   // int64 edge_index, [2, N_EDGES] flat
    const float*     W  = (const float*)d_in[2];
    const float*     b  = (const float*)d_in[3];
    float* out = (float*)d_out;

    // workspace layout (all sizes multiples of 16B; d_ws is 256B aligned)
    char* ws = (char*)d_ws;
    unsigned int*   bitmask = (unsigned int*)ws;                                  // 8 MiB
    float*          dinv    = (float*)(ws + (size_t)(8u << 20));                  // 32 KiB (padded to 64 KiB)
    unsigned short* xhi     = (unsigned short*)(ws + (size_t)(8u << 20) + (64u << 10)); // 8 MiB
    unsigned short* xlo     = xhi + (size_t)N_NODES * IN_C;                       // 8 MiB
    unsigned short* whi     = xlo + (size_t)N_NODES * IN_C;                       // 512 KiB
    unsigned short* wlo     = whi + (size_t)OUT_C * IN_C;                         // 512 KiB
    float*          hs      = (float*)(wlo + (size_t)OUT_C * IN_C);               // 16 MiB

    // 1. clear bitmask: 2,097,152 u32 = 524,288 uint4
    zero_bitmask_kernel<<<2048, 256, 0, stream>>>((uint4*)bitmask);

    // 2. scatter edge bits + self loops
    {
        const int total = N_EDGES + N_NODES;
        set_edges_kernel<<<(total + 255) / 256, 256, 0, stream>>>(ei, bitmask);
    }

    // 3. degrees -> deg^{-1/2}
    degree_kernel<<<(N_NODES + 255) / 256, 256, 0, stream>>>(bitmask, dinv);

    // 4. split x and W into bf16 hi/lo
    split_bf16_kernel<<<(N_NODES * IN_C + 255) / 256, 256, 0, stream>>>(x, xhi, xlo, N_NODES * IN_C);
    split_bf16_kernel<<<(OUT_C * IN_C + 255) / 256, 256, 0, stream>>>(W, whi, wlo, OUT_C * IN_C);

    // 5. WMMA GEMM with fused bias + dinv scaling: hs = dinv .* (x @ W^T + b)
    //    4096 wave-tiles (16x64), 8 waves/block -> 512 blocks
    gemm_wmma_kernel<<<512, 256, 0, stream>>>(xhi, xlo, whi, wlo, b, dinv, hs);

    // 6. sparse aggregation: out = dinv .* (A @ hs)
    aggregate_kernel<<<N_NODES, 256, 0, stream>>>(bitmask, hs, dinv, out);
}
